// MyAttention_31653908971626
// MI455X (gfx1250) — compile-verified, matrix operand order checked by reading, not run on hardware
//
#include <hip/hip_runtime.h>

typedef __attribute__((ext_vector_type(16))) __bf16 v16bf;
typedef __attribute__((ext_vector_type(4)))  __bf16 v4bf;
typedef __attribute__((ext_vector_type(8)))  float  v8f;

// ---- workspace layout (bytes) ----
#define WS_V     ((size_t)0)          // 16*256*4096 f32 = 67,108,864
#define WS_QK    ((size_t)67108864)   // 16*64*4096 f32  = 16,777,216
#define WS_WFRAG ((size_t)83886080)   // 20*8*2*512 bf16 =    327,680
#define WS_BIAS  ((size_t)84213760)   // 320 f32         =      1,280
#define WS_E     ((size_t)84215040)   // 128 f32 (eH|eW)
#define WS_ATT   ((size_t)84215552)   // 128 f32

// K-index inside a 16-bit A/B WMMA fragment (ISA 7.12.2):
// lane<16 holds K = j (j<8) and K = j+8 (j>=8, i.e. 16..23)
// lane>=16 holds K = j+8 and K = j+16.
__device__ __forceinline__ int frag_k(int j, int half) {
    return (j < 8) ? (j + 8 * half) : (j + 8 + 8 * half);
}

// ---------------------------------------------------------------------------
// K0a: pre-swizzle concatenated weights [wq;wk;wv] (320x256 f32) into per-lane
// bf16 hi/lo A-fragments: layout [mtile(20)][kstep(8)][hi/lo][lane(32)][16]
// ---------------------------------------------------------------------------
__global__ void pack_weights(const float* __restrict__ wq,
                             const float* __restrict__ wk,
                             const float* __restrict__ wv,
                             __bf16* __restrict__ wfrag) {
    const int ks   = blockIdx.x;        // 0..7   (K step of 32)
    const int mt   = blockIdx.y;        // 0..19  (M tile of 16)
    const int lane = threadIdx.x;       // 0..31
    const int half = lane >> 4;
    const int m    = mt * 16 + (lane & 15);
    __bf16* dst = wfrag + (size_t)(mt * 8 + ks) * 2 * 512 + lane * 16;
    #pragma unroll
    for (int j = 0; j < 16; ++j) {
        const int c = ks * 32 + frag_k(j, half);
        float f;
        if (m < 32)      f = wq[m * 256 + c];
        else if (m < 64) f = wk[(m - 32) * 256 + c];
        else             f = wv[(m - 64) * 256 + c];
        const __bf16 h = (__bf16)f;                 // high part
        const __bf16 l = (__bf16)(f - (float)h);    // residual
        dst[j]       = h;
        dst[512 + j] = l;
    }
}

__global__ void pack_bias(const float* __restrict__ bq,
                          const float* __restrict__ bk,
                          const float* __restrict__ bv,
                          float* __restrict__ bias) {
    const int t = threadIdx.x;
    if (t < 32)       bias[t] = bq[t];
    else if (t < 64)  bias[t] = bk[t - 32];
    else if (t < 320) bias[t] = bv[t - 64];
}

// ---------------------------------------------------------------------------
// K1: fused QKV GEMM. Per block (256 thr = 8 waves): batch b, M-supertile of
// 64 rows, N-supertile of 128 pixels.
// Staging converts x (f32) -> bf16 hi/lo ONCE and stores LDS directly in the
// WMMA B-fragment layout [ntile(8)][lane(32)][j(16)], so the hot loop is pure
// ds_load_b128 + v_wmma (no per-wave cvt, no scalar LDS reads).
// Each staging thread owns a 4(K)x4(p) block: 4 coalesced float4 row loads,
// and the 4 K-values are consecutive j's of one lane -> one 8B v4bf store per
// p per matrix.
// ---------------------------------------------------------------------------
__global__ __launch_bounds__(256) void gemm_qkv(
        const float* __restrict__ x, const __bf16* __restrict__ wfrag,
        const float* __restrict__ bias,
        float* __restrict__ qk, float* __restrict__ v) {
    __shared__ __bf16 bh[8][32][16];   // 8 KB  (hi fragments)
    __shared__ __bf16 bl[8][32][16];   // 8 KB  (lo fragments)

    const int tid  = threadIdx.x;
    const int wave = tid >> 5;
    const int lane = tid & 31;
    const int half = lane >> 4;
    const int ln   = lane & 15;

    const int b  = blockIdx.z;
    const int p0 = blockIdx.x * 128;
    const int mt = blockIdx.y * 4 + (wave & 3);   // global M tile 0..19
    const int nh = wave >> 2;                      // N half (0/1) of 64 px

    // staging decomposition: 8 k-quads x 32 p-quads
    const int kq  = (tid >> 5) * 4;                // 0,4,...,28
    const int pbt = (tid & 31) * 4;                // 0..124
    const int half2 = (kq >> 3) & 1;               // fragment half for rows kq..kq+3
    const int j0    = (kq & 7) + ((kq >> 4) << 3); // consecutive j slots

    v8f acc[4] = {};

    for (int ks = 0; ks < 8; ++ks) {
        // ---- stage + convert x[b, ks*32 + (kq..kq+3), p0+pbt..+3] ----
        const float* src = x + ((size_t)b * 256 + ks * 32 + kq) * 4096 + p0 + pbt;
        const float4 r0 = *(const float4*)(src);
        const float4 r1 = *(const float4*)(src + 4096);
        const float4 r2 = *(const float4*)(src + 8192);
        const float4 r3 = *(const float4*)(src + 12288);
        if (ks < 7) __builtin_prefetch(src + 32 * 4096, 0, 0);  // next K slab

        const float* f0 = &r0.x;
        const float* f1 = &r1.x;
        const float* f2 = &r2.x;
        const float* f3 = &r3.x;
        #pragma unroll
        for (int i = 0; i < 4; ++i) {
            const int p     = pbt + i;
            const int nt    = p >> 4;
            const int lane1 = (p & 15) + (half2 << 4);
            v4bf h, l;
            h[0] = (__bf16)f0[i]; l[0] = (__bf16)(f0[i] - (float)h[0]);
            h[1] = (__bf16)f1[i]; l[1] = (__bf16)(f1[i] - (float)h[1]);
            h[2] = (__bf16)f2[i]; l[2] = (__bf16)(f2[i] - (float)h[2]);
            h[3] = (__bf16)f3[i]; l[3] = (__bf16)(f3[i] - (float)h[3]);
            *(v4bf*)&bh[nt][lane1][j0] = h;
            *(v4bf*)&bl[nt][lane1][j0] = l;
        }
        __syncthreads();

        // ---- A fragments (pre-swizzled, one 32B load each) ----
        const __bf16* abase = wfrag + (size_t)(mt * 8 + ks) * 2 * 512 + lane * 16;
        const v16bf ahi = *(const v16bf*)(abase);
        const v16bf alo = *(const v16bf*)(abase + 512);

        #pragma unroll
        for (int nt = 0; nt < 4; ++nt) {
            const v16bf Bh = *(const v16bf*)&bh[nh * 4 + nt][lane][0];
            const v16bf Bl = *(const v16bf*)&bl[nh * 4 + nt][lane][0];
            acc[nt] = __builtin_amdgcn_wmma_f32_16x16x32_bf16(
                          false, ahi, false, Bh, (short)0, acc[nt], false, false);
            acc[nt] = __builtin_amdgcn_wmma_f32_16x16x32_bf16(
                          false, ahi, false, Bl, (short)0, acc[nt], false, false);
            acc[nt] = __builtin_amdgcn_wmma_f32_16x16x32_bf16(
                          false, alo, false, Bh, (short)0, acc[nt], false, false);
        }
        __syncthreads();
    }

    // ---- epilogue: + bias, scatter to qk (o<64) or v buffers ----
    const int o_base = mt * 16;
    #pragma unroll
    for (int nt = 0; nt < 4; ++nt) {
        const int p = p0 + nh * 64 + nt * 16 + ln;
        if (o_base < 64) {
            #pragma unroll
            for (int r = 0; r < 8; ++r) {
                const int o = o_base + half * 8 + r;
                qk[((size_t)b * 64 + o) * 4096 + p] = acc[nt][r] + bias[o];
            }
        } else {
            #pragma unroll
            for (int r = 0; r < 8; ++r) {
                const int o = o_base + half * 8 + r;
                v[((size_t)b * 256 + (o - 64)) * 4096 + p] = acc[nt][r] + bias[o];
            }
        }
    }
}

// ---------------------------------------------------------------------------
// K2: eH/eW logits: 128 dots of length 131072 (q rows / k rows are contiguous
// slabs in the qk buffer). One block per dot, float4 + LDS tree reduction.
// ---------------------------------------------------------------------------
__global__ __launch_bounds__(256) void pair_dots(const float* __restrict__ qk,
                                                 float* __restrict__ e) {
    __shared__ float red[256];
    const int eid  = blockIdx.x;        // 0..63 eH, 64..127 eW
    const int kind = eid >> 6;
    const int idx  = eid & 63;
    const int h = idx >> 4, w = (idx >> 2) & 3, g = idx & 3;
    const int bq = 4 * h + w;
    const int bk = (kind == 0) ? (4 * g + w) : (4 * h + g);
    const float4* qp = (const float4*)(qk + (size_t)bq * 64 * 4096);
    const float4* kp = (const float4*)(qk + ((size_t)bk * 64 + 32) * 4096);
    float s = 0.f;
    for (int i = threadIdx.x; i < 32768; i += 256) {
        const float4 a = qp[i], c = kp[i];
        s += a.x * c.x + a.y * c.y + a.z * c.z + a.w * c.w;
    }
    red[threadIdx.x] = s;
    __syncthreads();
    for (int st = 128; st > 0; st >>= 1) {
        if (threadIdx.x < st) red[threadIdx.x] += red[threadIdx.x + st];
        __syncthreads();
    }
    if (threadIdx.x == 0) e[eid] = red[0];
}

// ---------------------------------------------------------------------------
// K3: 16 softmaxes over 8 logits (eH diag masked to -inf)
// ---------------------------------------------------------------------------
__global__ void softmax16(const float* __restrict__ e, float* __restrict__ att) {
    const int t = threadIdx.x;
    if (t >= 16) return;
    const int h = t >> 2;
    float lg[8];
    #pragma unroll
    for (int g = 0; g < 4; ++g)
        lg[g] = (g == h) ? -__builtin_inff() : e[t * 4 + g];
    #pragma unroll
    for (int g = 0; g < 4; ++g) lg[4 + g] = e[64 + t * 4 + g];
    float m = lg[0];
    #pragma unroll
    for (int j = 1; j < 8; ++j) m = fmaxf(m, lg[j]);
    float ex[8], s = 0.f;
    #pragma unroll
    for (int j = 0; j < 8; ++j) { ex[j] = __expf(lg[j] - m); s += ex[j]; }
    const float inv = 1.f / s;
    #pragma unroll
    for (int j = 0; j < 8; ++j) att[t * 8 + j] = ex[j] * inv;
}

// ---------------------------------------------------------------------------
// K4: out = gamma * (sum_g aH[g]*v[4g+w] + sum_g aW[g]*v[4h+g]) + x
// float4 streaming; v+x live in the 192MB L2 so the 8x v reuse is cheap.
// ---------------------------------------------------------------------------
__global__ __launch_bounds__(256) void fuse_out(
        const float* __restrict__ x, const float* __restrict__ v,
        const float* __restrict__ att, const float* __restrict__ gamma,
        float* __restrict__ out) {
    const int b = blockIdx.y;
    const int h = b >> 2, w = b & 3;
    const int i4 = blockIdx.x * 256 + threadIdx.x;     // 0..262143 float4/batch
    const float g0 = gamma[0];
    float aH[4], aW[4];
    #pragma unroll
    for (int g = 0; g < 4; ++g) { aH[g] = att[b * 8 + g]; aW[g] = att[b * 8 + 4 + g]; }

    const float4 xv = ((const float4*)x)[(size_t)b * 262144 + i4];
    float sx = 0.f, sy = 0.f, sz = 0.f, sw = 0.f;
    #pragma unroll
    for (int g = 0; g < 4; ++g) {
        const float4 vh = ((const float4*)v)[(size_t)(4 * g + w) * 262144 + i4];
        sx += aH[g] * vh.x; sy += aH[g] * vh.y; sz += aH[g] * vh.z; sw += aH[g] * vh.w;
        const float4 vw = ((const float4*)v)[(size_t)(4 * h + g) * 262144 + i4];
        sx += aW[g] * vw.x; sy += aW[g] * vw.y; sz += aW[g] * vw.z; sw += aW[g] * vw.w;
    }
    float4 o;
    o.x = g0 * sx + xv.x; o.y = g0 * sy + xv.y;
    o.z = g0 * sz + xv.z; o.w = g0 * sw + xv.w;
    ((float4*)out)[(size_t)b * 262144 + i4] = o;
}

// ---------------------------------------------------------------------------
extern "C" void kernel_launch(void* const* d_in, const int* in_sizes, int n_in,
                              void* d_out, int out_size, void* d_ws, size_t ws_size,
                              hipStream_t stream) {
    const float* x     = (const float*)d_in[0];
    const float* wq    = (const float*)d_in[1];
    const float* bq    = (const float*)d_in[2];
    const float* wk    = (const float*)d_in[3];
    const float* bk    = (const float*)d_in[4];
    const float* wv    = (const float*)d_in[5];
    const float* bv    = (const float*)d_in[6];
    const float* gamma = (const float*)d_in[7];

    char* ws = (char*)d_ws;
    float*  ws_v     = (float*)(ws + WS_V);
    float*  ws_qk    = (float*)(ws + WS_QK);
    __bf16* ws_wfrag = (__bf16*)(ws + WS_WFRAG);
    float*  ws_bias  = (float*)(ws + WS_BIAS);
    float*  ws_e     = (float*)(ws + WS_E);
    float*  ws_att   = (float*)(ws + WS_ATT);
    float*  out      = (float*)d_out;

    pack_weights<<<dim3(8, 20), 32, 0, stream>>>(wq, wk, wv, ws_wfrag);
    pack_bias<<<1, 320, 0, stream>>>(bq, bk, bv, ws_bias);
    gemm_qkv<<<dim3(32, 5, 16), 256, 0, stream>>>(x, ws_wfrag, ws_bias, ws_qk, ws_v);
    pair_dots<<<128, 256, 0, stream>>>(ws_qk, ws_e);
    softmax16<<<1, 32, 0, stream>>>(ws_e, ws_att);
    fuse_out<<<dim3(1024, 16), 256, 0, stream>>>(x, ws_v, ws_att, gamma, out);
}
